// RealSchNetModel_42760694399567
// MI455X (gfx1250) — matrix operand order repaired and checked by reading, not compiled, not run on hardware
//
#include <hip/hip_runtime.h>
#include <hip/hip_bf16.h>
#include <math.h>

#define N_ATOMS 20000
#define N_EDGES 400000
#define NUM_GRAPHS 16
#define HIDDEN 128
#define GAUSS 50
#define GPAD 64
#define INTERACTIONS 6

typedef __attribute__((ext_vector_type(16))) __bf16 v16bf;
typedef __attribute__((ext_vector_type(8)))  float  v8f;

// ---------- helpers ----------

// native f32 -> bf16 convert (RNE); lowers to hardware cvt on gfx1250
static __device__ __forceinline__ __bf16 f2bfh(float f) {
  return (__bf16)f;
}
// shifted softplus: log(1+exp(x)) - log(2), numerically stable
static __device__ __forceinline__ float ssp(float x) {
  float sp = fmaxf(x, 0.f) + __logf(1.f + __expf(-fabsf(x)));
  return sp - 0.69314718055994531f;
}

union AFrag { v16bf v; uint4 q[2]; __bf16 h[16]; };
union BFrag { v16bf v; uint4 q[2]; };

static __device__ __forceinline__ v8f wmma_bf16(v16bf a, v16bf b, v8f c) {
  return __builtin_amdgcn_wmma_f32_16x16x32_bf16(false, a, false, b, (short)0, c,
                                                 false, false);
}

// B fragment from pre-swizzled buffer: each (nt,kt) tile is 32 lanes x 16 bf16,
// lane-major; per-lane 32 contiguous bytes.
static __device__ __forceinline__ v16bf load_bfrag(const __bf16* base, int nt,
                                                   int kt, int KT, int lane) {
  const uint4* p = (const uint4*)(base + ((((size_t)nt * KT + kt) * 32 + lane) << 4));
  BFrag f;
  f.q[0] = p[0];
  f.q[1] = p[1];
  return f.v;
}

// A fragment (16x32 bf16, M x K) from a row-major bf16 row pointer.
// lane<16 (hi=0): K = {kb..kb+7, kb+16..kb+23}; lane>=16 (hi=1): +8.
static __device__ __forceinline__ v16bf load_afrag_bf16(const __bf16* row, int kb,
                                                        int hi) {
  AFrag f;
  f.q[0] = *(const uint4*)(row + kb + hi * 8);
  f.q[1] = *(const uint4*)(row + kb + 16 + hi * 8);
  return f.v;
}

// A fragment built from a row-major f32 row (converted to bf16 in-register).
static __device__ __forceinline__ v16bf load_afrag_f32(const float* row, int kb,
                                                       int hi) {
  AFrag f;
  const float4* p0 = (const float4*)(row + kb + hi * 8);
  float4 a = p0[0], b = p0[1];
  const float4* p1 = (const float4*)(row + kb + 16 + hi * 8);
  float4 c = p1[0], d = p1[1];
  f.h[0] = f2bfh(a.x);  f.h[1] = f2bfh(a.y);  f.h[2] = f2bfh(a.z);  f.h[3] = f2bfh(a.w);
  f.h[4] = f2bfh(b.x);  f.h[5] = f2bfh(b.y);  f.h[6] = f2bfh(b.z);  f.h[7] = f2bfh(b.w);
  f.h[8] = f2bfh(c.x);  f.h[9] = f2bfh(c.y);  f.h[10] = f2bfh(c.z); f.h[11] = f2bfh(c.w);
  f.h[12] = f2bfh(d.x); f.h[13] = f2bfh(d.y); f.h[14] = f2bfh(d.z); f.h[15] = f2bfh(d.w);
  return f.v;
}

// ---------- prep kernels ----------

// h[n,c] = embedding[z[n], c]
__global__ void k_embed(const int* __restrict__ z, const float* __restrict__ emb,
                        float* __restrict__ h) {
  int idx = blockIdx.x * 256 + threadIdx.x;
  if (idx >= N_ATOMS * HIDDEN) return;
  int n = idx >> 7, c = idx & 127;
  h[idx] = emb[z[n] * HIDDEN + c];
}

// per-edge distance -> gaussian RBF (bf16, padded 50->64) + cosine cutoff C
__global__ void k_edge_rbf(const int* __restrict__ ei, const float* __restrict__ pos,
                           __bf16* __restrict__ rbf, float* __restrict__ Cb) {
  int e = blockIdx.x * 256 + threadIdx.x;
  if (e >= N_EDGES) return;
  int s = ei[e], d = ei[N_EDGES + e];
  float dx = pos[s * 3 + 0] - pos[d * 3 + 0];
  float dy = pos[s * 3 + 1] - pos[d * 3 + 1];
  float dz = pos[s * 3 + 2] - pos[d * 3 + 2];
  float dist = sqrtf(dx * dx + dy * dy + dz * dz);
  Cb[e] = 0.5f * (__cosf(dist * 0.31415926535f) + 1.0f);  // pi/10
  const float delta = 10.0f / 49.0f;
  const float coeff = -0.5f / (delta * delta);
  union { __bf16 h[GPAD]; uint4 q[GPAD / 8]; } t;
#pragma unroll
  for (int k = 0; k < GPAD; ++k) {
    float v = 0.f;
    if (k < GAUSS) {
      float r = dist - (float)k * delta;
      v = __expf(coeff * r * r);
    }
    t.h[k] = f2bfh(v);
  }
  uint4* dst = (uint4*)(rbf + (size_t)e * GPAD);
#pragma unroll
  for (int i = 0; i < GPAD / 8; ++i) dst[i] = t.q[i];
}

// swizzle a [Ksrc,128] f32 weight (row-major, W.T convention) into the WMMA
// B-fragment lane layout in bf16, K padded to Kpad (zeros).
__global__ void k_swizzle(const float* __restrict__ src, __bf16* __restrict__ dst,
                          int Ksrc, int Kpad) {
  int idx = blockIdx.x * 256 + threadIdx.x;
  if (idx >= Kpad * 128) return;
  int k = idx >> 7, n = idx & 127;
  float v = (k < Ksrc) ? src[k * 128 + n] : 0.f;
  int kt = k >> 5, kr = k & 31, nt = n >> 4;
  int lane = (n & 15) | ((kr & 8) << 1);   // cols on lanes 0-15, K bit 3 selects lanes 16-31
  int pos = (kr & 7) | ((kr & 16) >> 1);
  int KT = Kpad >> 5;
  dst[((((size_t)nt * KT + kt) * 32 + lane) << 4) + pos] = f2bfh(v);
}

// ---------- node GEMM: x = h @ lin1 (no bias) ----------
__global__ void __launch_bounds__(128) k_lin1(const float* __restrict__ h,
                                              const __bf16* __restrict__ wsw,
                                              float* __restrict__ x) {
  int lane = threadIdx.x & 31, wave = threadIdx.x >> 5;
  int tile = blockIdx.x * 4 + wave;
  int r0 = tile * 16;
  int l15 = lane & 15, hi = lane >> 4;
  int arow = r0 + l15;
  if (arow >= N_ATOMS) arow = N_ATOMS - 1;
  const v8f z8 = {0.f, 0.f, 0.f, 0.f, 0.f, 0.f, 0.f, 0.f};
  v8f acc[8];
#pragma unroll
  for (int nt = 0; nt < 8; ++nt) acc[nt] = z8;
#pragma unroll
  for (int kt = 0; kt < 4; ++kt) {
    v16bf a = load_afrag_f32(h + (size_t)arow * 128, kt * 32, hi);
#pragma unroll
    for (int nt = 0; nt < 8; ++nt)
      acc[nt] = wmma_bf16(a, load_bfrag(wsw, nt, kt, 4, lane), acc[nt]);
  }
#pragma unroll
  for (int nt = 0; nt < 8; ++nt)
#pragma unroll
    for (int v = 0; v < 8; ++v) {
      int row = r0 + v + 8 * hi;
      if (row < N_ATOMS) x[(size_t)row * 128 + nt * 16 + l15] = acc[nt][v];
    }
}

// ---------- fused edge kernel ----------
// per 16-edge tile: W = ssp(rbf@w1+b1)@w2+b2, W *= C; msg = x[src]*W;
// atomicAdd into agg[dst].
__global__ void __launch_bounds__(128) k_edge_conv(
    const __bf16* __restrict__ rbf, const float* __restrict__ Cb,
    const int* __restrict__ ei, const float* __restrict__ x,
    float* __restrict__ agg, const __bf16* __restrict__ w1sw,
    const __bf16* __restrict__ w2sw, const float* __restrict__ b1,
    const float* __restrict__ b2) {
  __shared__ float cts[4][16];
  __shared__ __align__(16) float scrf[4][2048];  // 8KB per wave, bf16/f32 phases
  int lane = threadIdx.x & 31, wave = threadIdx.x >> 5;
  int tile = blockIdx.x * 4 + wave;
  int e0 = tile * 16;
  int l15 = lane & 15, hi = lane >> 4;
  __bf16* scrb = (__bf16*)&scrf[wave][0];
  float* sw = &scrf[wave][0];
  if (lane < 16) cts[wave][lane] = Cb[e0 + lane];

  const v8f z8 = {0.f, 0.f, 0.f, 0.f, 0.f, 0.f, 0.f, 0.f};
  v8f acc[8];
#pragma unroll
  for (int nt = 0; nt < 8; ++nt) acc[nt] = z8;

  // stage 1: rbf[16,64] @ w1[64,128]
#pragma unroll
  for (int kt = 0; kt < 2; ++kt) {
    v16bf a = load_afrag_bf16(rbf + (size_t)(e0 + l15) * GPAD, kt * 32, hi);
#pragma unroll
    for (int nt = 0; nt < 8; ++nt)
      acc[nt] = wmma_bf16(a, load_bfrag(w1sw, nt, kt, 2, lane), acc[nt]);
  }
  // ssp(+b1) -> LDS scratch as bf16 (A layout source for stage 2)
#pragma unroll
  for (int nt = 0; nt < 8; ++nt) {
    float bias = b1[nt * 16 + l15];
#pragma unroll
    for (int v = 0; v < 8; ++v)
      scrb[(v + 8 * hi) * 128 + nt * 16 + l15] = f2bfh(ssp(acc[nt][v] + bias));
  }
  __syncthreads();

  // stage 2: t1[16,128] @ w2[128,128]
#pragma unroll
  for (int nt = 0; nt < 8; ++nt) acc[nt] = z8;
#pragma unroll
  for (int kt = 0; kt < 4; ++kt) {
    v16bf a = load_afrag_bf16(scrb + l15 * 128, kt * 32, hi);
#pragma unroll
    for (int nt = 0; nt < 8; ++nt)
      acc[nt] = wmma_bf16(a, load_bfrag(w2sw, nt, kt, 4, lane), acc[nt]);
  }
  __syncthreads();

  // W = (D + b2) * C  -> float scratch
#pragma unroll
  for (int nt = 0; nt < 8; ++nt) {
    float bias = b2[nt * 16 + l15];
#pragma unroll
    for (int v = 0; v < 8; ++v) {
      int m = v + 8 * hi;
      sw[m * 128 + nt * 16 + l15] = (acc[nt][v] + bias) * cts[wave][m];
    }
  }
  __syncthreads();

  // gather x[src], modulate, scatter-add to agg[dst]
  for (int e = 0; e < 16; ++e) {
    int s = ei[e0 + e];
    int dgt = ei[N_EDGES + e0 + e];
    const float* xs = x + (size_t)s * 128;
    float* ad = agg + (size_t)dgt * 128;
#pragma unroll
    for (int c = lane; c < 128; c += 32) atomicAdd(&ad[c], xs[c] * sw[e * 128 + c]);
  }
}

// ---------- node tail: h += (ssp(agg@lin2+b2)) @ blk + bb ----------
__global__ void __launch_bounds__(128) k_tail(const float* __restrict__ agg,
                                              float* __restrict__ h,
                                              const __bf16* __restrict__ w2sw,
                                              const float* __restrict__ b2,
                                              const __bf16* __restrict__ wbsw,
                                              const float* __restrict__ bb) {
  __shared__ __align__(16) __bf16 scrb_all[4][2048];
  int lane = threadIdx.x & 31, wave = threadIdx.x >> 5;
  int tile = blockIdx.x * 4 + wave;
  int r0 = tile * 16;
  int l15 = lane & 15, hi = lane >> 4;
  __bf16* scrb = &scrb_all[wave][0];
  int arow = r0 + l15;
  if (arow >= N_ATOMS) arow = N_ATOMS - 1;

  const v8f z8 = {0.f, 0.f, 0.f, 0.f, 0.f, 0.f, 0.f, 0.f};
  v8f acc[8];
#pragma unroll
  for (int nt = 0; nt < 8; ++nt) acc[nt] = z8;
#pragma unroll
  for (int kt = 0; kt < 4; ++kt) {
    v16bf a = load_afrag_f32(agg + (size_t)arow * 128, kt * 32, hi);
#pragma unroll
    for (int nt = 0; nt < 8; ++nt)
      acc[nt] = wmma_bf16(a, load_bfrag(w2sw, nt, kt, 4, lane), acc[nt]);
  }
#pragma unroll
  for (int nt = 0; nt < 8; ++nt) {
    float bias = b2[nt * 16 + l15];
#pragma unroll
    for (int v = 0; v < 8; ++v)
      scrb[(v + 8 * hi) * 128 + nt * 16 + l15] = f2bfh(ssp(acc[nt][v] + bias));
  }
  __syncthreads();
#pragma unroll
  for (int nt = 0; nt < 8; ++nt) acc[nt] = z8;
#pragma unroll
  for (int kt = 0; kt < 4; ++kt) {
    v16bf a = load_afrag_bf16(scrb + l15 * 128, kt * 32, hi);
#pragma unroll
    for (int nt = 0; nt < 8; ++nt)
      acc[nt] = wmma_bf16(a, load_bfrag(wbsw, nt, kt, 4, lane), acc[nt]);
  }
#pragma unroll
  for (int nt = 0; nt < 8; ++nt) {
    float bias = bb[nt * 16 + l15];
#pragma unroll
    for (int v = 0; v < 8; ++v) {
      int row = r0 + v + 8 * hi;
      if (row < N_ATOMS) {
        size_t idx = (size_t)row * 128 + nt * 16 + l15;
        h[idx] += acc[nt][v] + bias;
      }
    }
  }
}

// ---------- readout ----------
__global__ void k_readout(const float* __restrict__ h, const int* __restrict__ batch,
                          const float* __restrict__ w1, const float* __restrict__ bo1,
                          const float* __restrict__ w2, const float* __restrict__ bo2,
                          float* __restrict__ outtmp) {
  __shared__ float hh[64];
  int a = blockIdx.x;
  int t = threadIdx.x;
  float s = 0.f;
  const float* hr = h + (size_t)a * 128;
  for (int k = 0; k < 128; ++k) s += hr[k] * w1[k * 64 + t];
  hh[t] = ssp(s + bo1[t]);
  __syncthreads();
  if (t < 3) {
    float p = bo2[t];
    for (int k = 0; k < 64; ++k) p += hh[k] * w2[k * 3 + t];
    atomicAdd(&outtmp[batch[a] * 3 + t], p);
  }
}

__global__ void k_final(const float* __restrict__ outtmp,
                        const float* __restrict__ mean,
                        const float* __restrict__ stdv, float* __restrict__ out) {
  int i = threadIdx.x;
  if (i < NUM_GRAPHS * 3) out[i] = outtmp[i] * stdv[i % 3] + mean[i % 3];
}

// ---------- host ----------

extern "C" void kernel_launch(void* const* d_in, const int* in_sizes, int n_in,
                              void* d_out, int out_size, void* d_ws, size_t ws_size,
                              hipStream_t stream) {
  (void)in_sizes; (void)n_in; (void)out_size; (void)ws_size;
  const int* z = (const int*)d_in[0];
  const float* pos = (const float*)d_in[1];
  const int* batch = (const int*)d_in[2];
  const int* ei = (const int*)d_in[3];
  const float* emb = (const float*)d_in[4];
  const float* mlp_w1 = (const float*)d_in[5];
  const float* mlp_b1 = (const float*)d_in[6];
  const float* mlp_w2 = (const float*)d_in[7];
  const float* mlp_b2 = (const float*)d_in[8];
  const float* lin1_w = (const float*)d_in[9];
  const float* lin2_w = (const float*)d_in[10];
  const float* lin2_b = (const float*)d_in[11];
  const float* blk_w = (const float*)d_in[12];
  const float* blk_b = (const float*)d_in[13];
  const float* o1w = (const float*)d_in[14];
  const float* o1b = (const float*)d_in[15];
  const float* o2w = (const float*)d_in[16];
  const float* o2b = (const float*)d_in[17];
  const float* tmean = (const float*)d_in[18];
  const float* tstd = (const float*)d_in[19];
  float* out = (float*)d_out;

  char* ws = (char*)d_ws;
  size_t off = 0;
  auto alloc = [&](size_t bytes) -> void* {
    void* p = ws + off;
    off = (off + bytes + 255) & ~(size_t)255;
    return p;
  };
  __bf16* rbf = (__bf16*)alloc((size_t)N_EDGES * GPAD * 2);
  float* Cb = (float*)alloc((size_t)N_EDGES * 4);
  float* h = (float*)alloc((size_t)N_ATOMS * 128 * 4);
  float* x = (float*)alloc((size_t)N_ATOMS * 128 * 4);
  float* agg = (float*)alloc((size_t)N_ATOMS * 128 * 4);
  float* outtmp = (float*)alloc(256);
  __bf16* w1sw = (__bf16*)alloc((size_t)INTERACTIONS * 64 * 128 * 2);
  __bf16* w2sw = (__bf16*)alloc((size_t)INTERACTIONS * 128 * 128 * 2);
  __bf16* l1sw = (__bf16*)alloc((size_t)INTERACTIONS * 128 * 128 * 2);
  __bf16* l2sw = (__bf16*)alloc((size_t)INTERACTIONS * 128 * 128 * 2);
  __bf16* blsw = (__bf16*)alloc((size_t)INTERACTIONS * 128 * 128 * 2);

  // weight swizzles (bf16, B-fragment layout)
  for (int t = 0; t < INTERACTIONS; ++t) {
    k_swizzle<<<(64 * 128 + 255) / 256, 256, 0, stream>>>(
        mlp_w1 + (size_t)t * GAUSS * 128, w1sw + (size_t)t * 64 * 128, GAUSS, 64);
    k_swizzle<<<(128 * 128 + 255) / 256, 256, 0, stream>>>(
        mlp_w2 + (size_t)t * 128 * 128, w2sw + (size_t)t * 128 * 128, 128, 128);
    k_swizzle<<<(128 * 128 + 255) / 256, 256, 0, stream>>>(
        lin1_w + (size_t)t * 128 * 128, l1sw + (size_t)t * 128 * 128, 128, 128);
    k_swizzle<<<(128 * 128 + 255) / 256, 256, 0, stream>>>(
        lin2_w + (size_t)t * 128 * 128, l2sw + (size_t)t * 128 * 128, 128, 128);
    k_swizzle<<<(128 * 128 + 255) / 256, 256, 0, stream>>>(
        blk_w + (size_t)t * 128 * 128, blsw + (size_t)t * 128 * 128, 128, 128);
  }

  k_embed<<<(N_ATOMS * 128 + 255) / 256, 256, 0, stream>>>(z, emb, h);
  k_edge_rbf<<<(N_EDGES + 255) / 256, 256, 0, stream>>>(ei, pos, rbf, Cb);

  const int nodeBlocks = (N_ATOMS / 16 + 3) / 4;   // 313 blocks of 4 waves
  const int edgeBlocks = N_EDGES / 16 / 4;         // 6250 blocks, exact

  for (int t = 0; t < INTERACTIONS; ++t) {
    k_lin1<<<nodeBlocks, 128, 0, stream>>>(h, l1sw + (size_t)t * 128 * 128, x);
    hipMemsetAsync(agg, 0, (size_t)N_ATOMS * 128 * 4, stream);
    k_edge_conv<<<edgeBlocks, 128, 0, stream>>>(
        rbf, Cb, ei, x, agg, w1sw + (size_t)t * 64 * 128,
        w2sw + (size_t)t * 128 * 128, mlp_b1 + (size_t)t * 128,
        mlp_b2 + (size_t)t * 128);
    k_tail<<<nodeBlocks, 128, 0, stream>>>(agg, h, l2sw + (size_t)t * 128 * 128,
                                           lin2_b + (size_t)t * 128,
                                           blsw + (size_t)t * 128 * 128,
                                           blk_b + (size_t)t * 128);
  }

  hipMemsetAsync(outtmp, 0, NUM_GRAPHS * 3 * sizeof(float), stream);
  k_readout<<<N_ATOMS, 64, 0, stream>>>(h, batch, o1w, o1b, o2w, o2b, outtmp);
  k_final<<<1, 64, 0, stream>>>(outtmp, tmean, tstd, out);
}
